// GraphAttention_38336878084772
// MI455X (gfx1250) — matrix operand order repaired and checked by reading, not compiled, not run on hardware
//
#include <hip/hip_runtime.h>
#include <math.h>

// ---------------------------------------------------------------------------
// MI455X (gfx1250) fused equivariant graph attention, bf16 WMMA everywhere.
// Weights are pre-packed per launch into transposed bf16 (WT[n][k]) so each
// lane's B fragment is two 16B loads (2x b128) with zero converts.
// Node vector channels are de-interleaved once so all A fragments use
// contiguous 128-bit loads.
// ---------------------------------------------------------------------------

typedef __bf16 bf16_t;
typedef __attribute__((ext_vector_type(16))) __bf16 v16bf;
typedef __attribute__((ext_vector_type(8)))  __bf16 v8bf;
typedef __attribute__((ext_vector_type(8)))  float  v8f;
typedef __attribute__((ext_vector_type(4)))  float  v4f;

constexpr int kN   = 10000;
constexpr int kE   = 160000;
constexpr int kN64 = kN * 64;

// ---------------- WMMA helpers (16x16x32 bf16, f32 accumulate) -------------

__device__ inline v8f wmma_bf16(v16bf a, v16bf b, v8f c) {
  return __builtin_amdgcn_wmma_f32_16x16x32_bf16(false, a, false, b, (short)0, c,
                                                 false, false);
}

// A fragment 16x32 (ISA 7.12.2): lane (m = l&15, h = l>>4) holds two
// contiguous 8-element chunks: k in [k0+8h, +8) and [k0+16+8h, +8).

// from bf16 buffer (LDS or global), local rows 0..15, 32B-aligned strides
__device__ inline v16bf a_frag_bf16(const bf16_t* A, int k0, int ld) {
  int l = threadIdx.x & 31, m = l & 15, h = l >> 4;
  const bf16_t* p = A + m * ld + k0 + 8 * h;
  v8bf lo = *(const v8bf*)p;
  v8bf hi = *(const v8bf*)(p + 16);
  return __builtin_shufflevector(lo, hi, 0, 1, 2, 3, 4, 5, 6, 7,
                                 8, 9, 10, 11, 12, 13, 14, 15);
}

// from contiguous f32 source (element stride 1)
__device__ inline v16bf a_frag_f32c(const float* A, int row0, int k0, int rs) {
  int l = threadIdx.x & 31, m = l & 15, h = l >> 4;
  const float* p = A + (size_t)(row0 + m) * rs + k0 + 8 * h;
  v4f q0 = *(const v4f*)(p);
  v4f q1 = *(const v4f*)(p + 4);
  v4f q2 = *(const v4f*)(p + 16);
  v4f q3 = *(const v4f*)(p + 20);
  v16bf a;
#pragma unroll
  for (int j = 0; j < 4; ++j) {
    a[j]      = (bf16_t)q0[j];
    a[4 + j]  = (bf16_t)q1[j];
    a[8 + j]  = (bf16_t)q2[j];
    a[12 + j] = (bf16_t)q3[j];
  }
  return a;
}

// B fragment 32x16 from packed transposed bf16 weights WT[n][k] (ld = K):
// lane (n = l&15, h = l>>4) reads k in [k0+8h, +8) and [k0+16+8h, +8).
__device__ inline v16bf b_frag(const bf16_t* WT, int k0, int n0, int K) {
  int l = threadIdx.x & 31, n = l & 15, h = l >> 4;
  const bf16_t* p = WT + (size_t)(n0 + n) * K + k0 + h * 8;
  v8bf lo = *(const v8bf*)p;
  v8bf hi = *(const v8bf*)(p + 16);
  return __builtin_shufflevector(lo, hi, 0, 1, 2, 3, 4, 5, 6, 7,
                                 8, 9, 10, 11, 12, 13, 14, 15);
}

// C/D store helpers. Lane n = lane&15, row m = r + 8*half.
__device__ inline void st_glb_f32(float* C, int row0, int n0, int ld, v8f acc,
                                  float scale, const float* __restrict__ bias) {
  int l = threadIdx.x & 31, n = l & 15, h = l >> 4;
  float b = bias[n0 + n];
#pragma unroll
  for (int r = 0; r < 8; ++r) {
    int m = r + 8 * h;
    C[(size_t)(row0 + m) * ld + n0 + n] = acc[r] * scale + b;
  }
}

__device__ inline void st_glb_f32_nb(float* C, int row0, int n0, int ld, v8f acc,
                                     float scale) {
  int l = threadIdx.x & 31, n = l & 15, h = l >> 4;
#pragma unroll
  for (int r = 0; r < 8; ++r) {
    int m = r + 8 * h;
    C[(size_t)(row0 + m) * ld + n0 + n] = acc[r] * scale;
  }
}

__device__ inline void st_lds_f32(float* C, int n0, int ld, v8f acc,
                                  float scale, const float* __restrict__ bias) {
  int l = threadIdx.x & 31, n = l & 15, h = l >> 4;
  float b = bias[n0 + n];
#pragma unroll
  for (int r = 0; r < 8; ++r) {
    int m = r + 8 * h;
    C[m * ld + n0 + n] = acc[r] * scale + b;
  }
}

__device__ inline void st_lds_bf16(bf16_t* C, int n0, int ld, v8f acc,
                                   float scale, const float* __restrict__ bias) {
  int l = threadIdx.x & 31, n = l & 15, h = l >> 4;
  float b = bias[n0 + n];
#pragma unroll
  for (int r = 0; r < 8; ++r) {
    int m = r + 8 * h;
    C[m * ld + n0 + n] = (bf16_t)(acc[r] * scale + b);
  }
}

__device__ inline void st_lds_silu_bf16(bf16_t* C, int n0, int ld, v8f acc,
                                        const float* __restrict__ bias) {
  int l = threadIdx.x & 31, n = l & 15, h = l >> 4;
  float b = bias[n0 + n];
#pragma unroll
  for (int r = 0; r < 8; ++r) {
    int m = r + 8 * h;
    float v = acc[r] + b;
    v = v / (1.f + expf(-v));  // silu
    C[m * ld + n0 + n] = (bf16_t)v;
  }
}

__device__ inline void atomicMaxFloat(float* addr, float val) {
  int* ai = reinterpret_cast<int*>(addr);
  int cur = __float_as_int(*addr);
  while (__int_as_float(cur) < val) {
    int prev = atomicCAS(ai, cur, __float_as_int(val));
    if (prev == cur) break;
    cur = prev;
  }
}

// ---------------- K-pack: W (KxN f32, row major) -> WT (NxK bf16) ----------
__global__ void k_pack_wt(const float* __restrict__ W, bf16_t* __restrict__ WT,
                          int K, int N) {
  int i = blockIdx.x * blockDim.x + threadIdx.x;
  if (i >= K * N) return;
  int n = i / K, k = i % K;
  WT[i] = (bf16_t)W[(size_t)k * N + n];
}

// ---------------- K-deint: node_input[:,128:] (i*3+c) -> v_in [c][n][64] ---
__global__ void k_deinterleave_v(const float* __restrict__ node_input,
                                 float* __restrict__ v_in) {
  int i = blockIdx.x * blockDim.x + threadIdx.x;
  if (i >= kN * 192) return;
  int c = i / kN64, r = i % kN64, n = r >> 6, k = r & 63;
  v_in[i] = node_input[(size_t)n * 320 + 128 + k * 3 + c];
}

// ---------------- K0: init accumulators ------------------------------------
// base: contiguous [node_s(N*128) | node_v(N*192) | amax(N*4) | denom(N*4)]
__global__ void k_init(float* base, int n) {
  int i = blockIdx.x * blockDim.x + threadIdx.x;
  if (i >= n) return;
  const int lo = kN * (128 + 192), hi = kN * (128 + 192 + 4);
  base[i] = (i >= lo && i < hi) ? -1e30f : 0.f;
}

// ---------------- K1: node input projections -------------------------------
__global__ __launch_bounds__(128) void k_node_proj(
    const float* __restrict__ node_input, const float* __restrict__ v_in,
    const bf16_t* __restrict__ wt_src0, const float* __restrict__ b_src0,
    const bf16_t* __restrict__ wt_src1,
    const bf16_t* __restrict__ wt_dst0, const float* __restrict__ b_dst0,
    const bf16_t* __restrict__ wt_dst1,
    float* __restrict__ s_src, float* __restrict__ s_dst,
    float* __restrict__ v_src, float* __restrict__ v_dst) {
  const int row0 = blockIdx.x * 16;
  const int wave = threadIdx.x >> 5;
  const float is128 = 0.08838834764831845f;  // 1/sqrt(128)
  const float is64  = 0.125f;                // 1/sqrt(64)

  for (int ct = wave; ct < 8; ct += 4) {
    v8f aS = {}, aD = {};
#pragma unroll
    for (int kk = 0; kk < 4; ++kk) {
      v16bf a  = a_frag_f32c(node_input, row0, kk * 32, 320);
      v16bf bs = b_frag(wt_src0, kk * 32, ct * 16, 128);
      v16bf bd = b_frag(wt_dst0, kk * 32, ct * 16, 128);
      aS = wmma_bf16(a, bs, aS);
      aD = wmma_bf16(a, bd, aD);
    }
    st_glb_f32(s_src, row0, ct * 16, 128, aS, is128, b_src0);
    st_glb_f32(s_dst, row0, ct * 16, 128, aD, is128, b_dst0);
  }
  for (int t = wave; t < 12; t += 4) {
    int c = t >> 2, ct = t & 3;
    v8f aS = {}, aD = {};
#pragma unroll
    for (int kk = 0; kk < 2; ++kk) {
      v16bf a  = a_frag_f32c(v_in + (size_t)c * kN64, row0, kk * 32, 64);
      v16bf bs = b_frag(wt_src1, kk * 32, ct * 16, 64);
      v16bf bd = b_frag(wt_dst1, kk * 32, ct * 16, 64);
      aS = wmma_bf16(a, bs, aS);
      aD = wmma_bf16(a, bd, aD);
    }
    st_glb_f32_nb(v_src + (size_t)c * kN64, row0, ct * 16, 64, aS, is64);
    st_glb_f32_nb(v_dst + (size_t)c * kN64, row0, ct * 16, 64, aD, is64);
  }
}

// ---------------- K2: fused per-edge kernel (16 edges / block) -------------
__global__ __launch_bounds__(128) void k_edge(
    const float* __restrict__ edge_attr, const float* __restrict__ edge_scalars,
    const int* __restrict__ esrc, const int* __restrict__ edst,
    const bf16_t* __restrict__ wt_rw1, const float* __restrict__ rb1,
    const bf16_t* __restrict__ wt_rw2, const float* __restrict__ rb2,
    const bf16_t* __restrict__ wt_rw3, const float* __restrict__ rb3,
    const bf16_t* __restrict__ wt_lw0, const float* __restrict__ lb0,
    const bf16_t* __restrict__ wt_lw1, const float* __restrict__ alpha_dot,
    const float* __restrict__ s_src, const float* __restrict__ s_dst,
    const float* __restrict__ v_src, const float* __restrict__ v_dst,
    float* __restrict__ alpha_out, float* __restrict__ amax,
    bf16_t* __restrict__ val_s_out, bf16_t* __restrict__ m1_out) {
  __shared__ bf16_t sh_h[16 * 64];
  __shared__ bf16_t sh_h2[16 * 64];
  __shared__ bf16_t sh_w[16 * 448];
  __shared__ bf16_t sh_ms[16 * 128];
  __shared__ float  sh_mv[16 * 192];  // [e][c][64]
  __shared__ bf16_t sh_d0[16 * 192];
  __shared__ bf16_t sh_d1[16 * 256];  // one component at a time
  __shared__ float  sh_m0[16 * 160];
  __shared__ float  sh_ev[16 * 3];
  __shared__ float  sh_es[16];
  __shared__ int    sh_src[16], sh_dst[16];

  const int e0 = blockIdx.x * 16;
  const int t = threadIdx.x;
  const int wave = t >> 5;

  // ---- stage A: edge metadata + gathers --------------------------------
  if (t < 16) {
    sh_src[t] = esrc[e0 + t];
    sh_dst[t] = edst[e0 + t];
    sh_es[t]  = edge_attr[(size_t)(e0 + t) * 4];
  }
  if (t < 48) {
    int e = t / 3, c = t % 3;
    sh_ev[e * 3 + c] = edge_attr[(size_t)(e0 + e) * 4 + 1 + c];
  }
  if (t == 0 && e0 + 16 < kE)
    __builtin_prefetch(edge_scalars + (size_t)(e0 + 16) * 64, 0, 1);
  __syncthreads();
  for (int i = t; i < 16 * 128; i += 128) {
    int e = i >> 7, k = i & 127;
    sh_ms[i] = (bf16_t)(s_src[(size_t)sh_src[e] * 128 + k] +
                        s_dst[(size_t)sh_dst[e] * 128 + k]);
  }
  for (int i = t; i < 16 * 192; i += 128) {
    int e = i / 192, r = i % 192, c = r >> 6, k = r & 63;
    sh_mv[i] = v_src[(size_t)c * kN64 + (size_t)sh_src[e] * 64 + k] +
               v_dst[(size_t)c * kN64 + (size_t)sh_dst[e] * 64 + k];
  }

  // ---- stage B: h1 = silu(edge_scalars @ rw1 + rb1) --------------------
  {
    int ct = wave;  // 4 tiles, one per wave
    v8f acc = {};
#pragma unroll
    for (int kk = 0; kk < 2; ++kk) {
      v16bf a = a_frag_f32c(edge_scalars, e0, kk * 32, 64);
      v16bf b = b_frag(wt_rw1, kk * 32, ct * 16, 64);
      acc = wmma_bf16(a, b, acc);
    }
    st_lds_silu_bf16(sh_h, ct * 16, 64, acc, rb1);
  }
  __syncthreads();

  // ---- stage C: h2 = silu(h1 @ rw2 + rb2) ------------------------------
  {
    int ct = wave;
    v8f acc = {};
#pragma unroll
    for (int kk = 0; kk < 2; ++kk) {
      v16bf a = a_frag_bf16(sh_h, kk * 32, 64);
      v16bf b = b_frag(wt_rw2, kk * 32, ct * 16, 64);
      acc = wmma_bf16(a, b, acc);
    }
    st_lds_silu_bf16(sh_h2, ct * 16, 64, acc, rb2);
  }
  __syncthreads();

  // ---- stage D: w = h2 @ rw3 + rb3  (16x448) ---------------------------
  for (int ct = wave; ct < 28; ct += 4) {
    v8f acc = {};
#pragma unroll
    for (int kk = 0; kk < 2; ++kk) {
      v16bf a = a_frag_bf16(sh_h2, kk * 32, 64);
      v16bf b = b_frag(wt_rw3, kk * 32, ct * 16, 64);
      acc = wmma_bf16(a, b, acc);
    }
    st_lds_bf16(sh_w, ct * 16, 448, acc, 1.f, rb3);
  }
  __syncthreads();

  // ---- stage E: d0 = [w_ss*ms*es | w_vv0*dot(mv,ev)/sqrt3] (16x192) ----
  for (int i = t; i < 16 * 192; i += 128) {
    int e = i / 192, k = i % 192;
    float val;
    if (k < 128) {
      val = (float)sh_w[e * 448 + k] * (float)sh_ms[e * 128 + k] * sh_es[e];
    } else {
      int k2 = k - 128;
      float dot = sh_mv[e * 192 + k2]       * sh_ev[e * 3 + 0] +
                  sh_mv[e * 192 + 64 + k2]  * sh_ev[e * 3 + 1] +
                  sh_mv[e * 192 + 128 + k2] * sh_ev[e * 3 + 2];
      val = (float)sh_w[e * 448 + 320 + k2] * dot * 0.5773502691896258f;
    }
    sh_d0[e * 192 + k] = (bf16_t)val;
  }
  __syncthreads();

  // ---- stage F: m0 = d0 @ lw0 / sqrt(192) + lb0  (16x160) --------------
  for (int ct = wave; ct < 10; ct += 4) {
    v8f acc = {};
#pragma unroll
    for (int kk = 0; kk < 6; ++kk) {
      v16bf a = a_frag_bf16(sh_d0, kk * 32, 192);
      v16bf b = b_frag(wt_lw0, kk * 32, ct * 16, 192);
      acc = wmma_bf16(a, b, acc);
    }
    st_lds_f32(sh_m0, ct * 16, 160, acc, 0.07216878364870323f, lb0);
  }
  __syncthreads();

  // ---- stage G: per component: d1_c then m1_c = d1_c @ lw1 / 16 --------
  for (int c = 0; c < 3; ++c) {
    int c1 = (c + 1) % 3, c2 = (c + 2) % 3;
    for (int i = t; i < 16 * 256; i += 128) {
      int e = i / 256, k = i % 256;
      float evc = sh_ev[e * 3 + c];
      float val;
      if (k < 128) {  // o1a: (w_sv*ms)*ev_c
        val = (float)sh_w[e * 448 + 128 + k] * (float)sh_ms[e * 128 + k] * evc;
      } else if (k < 192) {  // o1b: (w_vs*es)*mv_c
        int k2 = k - 128;
        val = (float)sh_w[e * 448 + 256 + k2] * sh_es[e] *
              sh_mv[e * 192 + c * 64 + k2];
      } else {  // o1c: w_vv1*cross(mv,ev)_c/sqrt2
        int k2 = k - 192;
        float cr = sh_mv[e * 192 + c1 * 64 + k2] * sh_ev[e * 3 + c2] -
                   sh_mv[e * 192 + c2 * 64 + k2] * sh_ev[e * 3 + c1];
        val = (float)sh_w[e * 448 + 384 + k2] * cr * 0.7071067811865476f;
      }
      sh_d1[e * 256 + k] = (bf16_t)val;
    }
    __syncthreads();
    {
      int ct = wave;  // 4 tiles, one per wave
      v8f acc = {};
#pragma unroll
      for (int kk = 0; kk < 8; ++kk) {
        v16bf a = a_frag_bf16(sh_d1, kk * 32, 256);
        v16bf b = b_frag(wt_lw1, kk * 32, ct * 16, 256);
        acc = wmma_bf16(a, b, acc);
      }
      int l = threadIdx.x & 31, n = l & 15, hh = l >> 4;
#pragma unroll
      for (int r = 0; r < 8; ++r) {
        int m = r + 8 * hh;
        m1_out[(size_t)(e0 + m) * 192 + c * 64 + ct * 16 + n] =
            (bf16_t)(acc[r] * 0.0625f);  // 1/sqrt(256)
      }
    }
    __syncthreads();
  }

  // ---- stage H: alpha logits + segment max, emit val_s -----------------
  if (t < 64) {
    int e = t >> 2, h = t & 3;
    float s = 0.f;
#pragma unroll
    for (int k = 0; k < 8; ++k) {
      float x = sh_m0[e * 160 + h * 40 + k];
      float sig = 1.f / (1.f + expf(-x));
      float xa = 0.6f * x + 0.4f * x * (2.f * sig - 1.f);
      s += xa * alpha_dot[h * 8 + k];
    }
    alpha_out[(size_t)(e0 + e) * 4 + h] = s;
    atomicMaxFloat(&amax[(size_t)sh_dst[e] * 4 + h], s);
  }
  for (int i = t; i < 16 * 128; i += 128) {
    int e = i >> 7, j = i & 127, h = j >> 5, k2 = j & 31;
    val_s_out[(size_t)(e0 + e) * 128 + j] = (bf16_t)sh_m0[e * 160 + h * 40 + 8 + k2];
  }
}

// ---------------- K3: softmax denominator ----------------------------------
__global__ void k_denom(float* __restrict__ alpha, const int* __restrict__ edst,
                        const float* __restrict__ amax, float* __restrict__ denom) {
  int i = blockIdx.x * blockDim.x + threadIdx.x;
  if (i >= kE * 4) return;
  int e = i >> 2, h = i & 3, d = edst[e];
  float v = expf(alpha[i] - amax[(size_t)d * 4 + h]);
  alpha[i] = v;  // becomes ealpha
  atomicAdd(&denom[(size_t)d * 4 + h], v);
}

// ---------------- K4: weighted scatter-aggregate (1 edge / block) ----------
__global__ __launch_bounds__(128) void k_aggregate(
    const float* __restrict__ ealpha, const int* __restrict__ edst,
    const float* __restrict__ denom, const bf16_t* __restrict__ val_s,
    const bf16_t* __restrict__ m1v, float* __restrict__ node_s,
    float* __restrict__ node_v) {
  const int e = blockIdx.x;
  __shared__ float aN[4];
  __shared__ int d;
  int t = threadIdx.x;
  if (t == 0) d = edst[e];
  __syncthreads();
  if (t < 4)
    aN[t] = ealpha[(size_t)e * 4 + t] / fmaxf(denom[(size_t)d * 4 + t], 1e-12f);
  __syncthreads();
  {
    float v = aN[t >> 5] * (float)val_s[(size_t)e * 128 + t];
    atomicAdd(&node_s[(size_t)d * 128 + t], v);
  }
  for (int j = t; j < 192; j += 128) {
    int c = j >> 6, i = j & 63;
    float v = aN[i >> 4] * (float)m1v[(size_t)e * 192 + j];
    atomicAdd(&node_v[(size_t)c * kN64 + (size_t)d * 64 + i], v);
  }
}

// ---------------- K5: output projections -----------------------------------
__global__ __launch_bounds__(128) void k_out_proj(
    const float* __restrict__ node_s, const float* __restrict__ node_v,
    const bf16_t* __restrict__ wt_pw0, const float* __restrict__ pb0,
    const bf16_t* __restrict__ wt_pw1, float* __restrict__ out) {
  const int row0 = blockIdx.x * 16;
  const int wave = threadIdx.x >> 5;
  const float is128 = 0.08838834764831845f;
  const float is64  = 0.125f;
  for (int ct = wave; ct < 8; ct += 4) {
    v8f acc = {};
#pragma unroll
    for (int kk = 0; kk < 4; ++kk) {
      v16bf a = a_frag_f32c(node_s, row0, kk * 32, 128);
      v16bf b = b_frag(wt_pw0, kk * 32, ct * 16, 128);
      acc = wmma_bf16(a, b, acc);
    }
    st_glb_f32(out, row0, ct * 16, 320, acc, is128, pb0);
  }
  for (int tk = wave; tk < 12; tk += 4) {
    int c = tk >> 2, ct = tk & 3;
    v8f acc = {};
#pragma unroll
    for (int kk = 0; kk < 2; ++kk) {
      v16bf a = a_frag_f32c(node_v + (size_t)c * kN64, row0, kk * 32, 64);
      v16bf b = b_frag(wt_pw1, kk * 32, ct * 16, 64);
      acc = wmma_bf16(a, b, acc);
    }
    int l = threadIdx.x & 31, n = l & 15, hh = l >> 4;
#pragma unroll
    for (int r = 0; r < 8; ++r) {
      int m = r + 8 * hh;
      out[(size_t)(row0 + m) * 320 + 128 + (ct * 16 + n) * 3 + c] = acc[r] * is64;
    }
  }
}

// ---------------------------------------------------------------------------
extern "C" void kernel_launch(void* const* d_in, const int* in_sizes, int n_in,
                              void* d_out, int out_size, void* d_ws, size_t ws_size,
                              hipStream_t stream) {
  const float* node_input   = (const float*)d_in[0];
  const float* edge_attr    = (const float*)d_in[1];
  const float* edge_scalars = (const float*)d_in[2];
  const int*   edge_src     = (const int*)d_in[3];
  const int*   edge_dst     = (const int*)d_in[4];
  const float* w_src0 = (const float*)d_in[5];
  const float* b_src0 = (const float*)d_in[6];
  const float* w_src1 = (const float*)d_in[7];
  const float* w_dst0 = (const float*)d_in[8];
  const float* b_dst0 = (const float*)d_in[9];
  const float* w_dst1 = (const float*)d_in[10];
  const float* rw1 = (const float*)d_in[11];
  const float* rb1 = (const float*)d_in[12];
  const float* rw2 = (const float*)d_in[13];
  const float* rb2 = (const float*)d_in[14];
  const float* rw3 = (const float*)d_in[15];
  const float* rb3 = (const float*)d_in[16];
  const float* lw0 = (const float*)d_in[17];
  const float* lb0 = (const float*)d_in[18];
  const float* lw1 = (const float*)d_in[19];
  const float* alpha_dot = (const float*)d_in[20];
  const float* pw0 = (const float*)d_in[21];
  const float* pb0 = (const float*)d_in[22];
  const float* pw1 = (const float*)d_in[23];

  // ---- workspace layout (floats unless noted); ~152 MB + 0.3 MB weights ---
  float* ws = (float*)d_ws;
  float* s_src  = ws; ws += (size_t)kN * 128;
  float* s_dst  = ws; ws += (size_t)kN * 128;
  float* v_src  = ws; ws += (size_t)kN * 192;
  float* v_dst  = ws; ws += (size_t)kN * 192;
  float* v_in   = ws; ws += (size_t)kN * 192;  // de-interleaved input v
  float* node_s = ws; ws += (size_t)kN * 128;  // ---- init region start
  float* node_v = ws; ws += (size_t)kN * 192;
  float* amax   = ws; ws += (size_t)kN * 4;
  float* denom  = ws; ws += (size_t)kN * 4;    // ---- init region end
  float* alpha  = ws; ws += (size_t)kE * 4;
  bf16_t* val_s = (bf16_t*)ws; ws += (size_t)kE * 64;   // E*128 bf16
  bf16_t* m1v   = (bf16_t*)ws; ws += (size_t)kE * 96;   // E*192 bf16
  // packed transposed bf16 weights (WT[n*K + k])
  bf16_t* wb = (bf16_t*)ws;
  bf16_t* wt_src0 = wb; wb += 128 * 128;
  bf16_t* wt_dst0 = wb; wb += 128 * 128;
  bf16_t* wt_src1 = wb; wb += 64 * 64;
  bf16_t* wt_dst1 = wb; wb += 64 * 64;
  bf16_t* wt_rw1  = wb; wb += 64 * 64;
  bf16_t* wt_rw2  = wb; wb += 64 * 64;
  bf16_t* wt_rw3  = wb; wb += 448 * 64;   // N=448, K=64
  bf16_t* wt_lw0  = wb; wb += 160 * 192;  // N=160, K=192
  bf16_t* wt_lw1  = wb; wb += 64 * 256;   // N=64,  K=256
  bf16_t* wt_pw0  = wb; wb += 128 * 128;
  bf16_t* wt_pw1  = wb; wb += 64 * 64;

  // ---- pack weights + de-interleave v (tiny; graph-captured) --------------
  auto pack = [&](const float* W, bf16_t* WT, int K, int N) {
    int n = K * N;
    k_pack_wt<<<(n + 255) / 256, 256, 0, stream>>>(W, WT, K, N);
  };
  pack(w_src0, wt_src0, 128, 128);
  pack(w_dst0, wt_dst0, 128, 128);
  pack(w_src1, wt_src1, 64, 64);
  pack(w_dst1, wt_dst1, 64, 64);
  pack(rw1, wt_rw1, 64, 64);
  pack(rw2, wt_rw2, 64, 64);
  pack(rw3, wt_rw3, 64, 448);
  pack(lw0, wt_lw0, 192, 160);
  pack(lw1, wt_lw1, 256, 64);
  pack(pw0, wt_pw0, 128, 128);
  pack(pw1, wt_pw1, 64, 64);

  k_deinterleave_v<<<(kN * 192 + 255) / 256, 256, 0, stream>>>(node_input, v_in);

  const int initN = kN * (128 + 192 + 4 + 4);
  k_init<<<(initN + 255) / 256, 256, 0, stream>>>(node_s, initN);

  k_node_proj<<<kN / 16, 128, 0, stream>>>(node_input, v_in,
                                           wt_src0, b_src0, wt_src1,
                                           wt_dst0, b_dst0, wt_dst1,
                                           s_src, s_dst, v_src, v_dst);

  k_edge<<<kE / 16, 128, 0, stream>>>(edge_attr, edge_scalars, edge_src, edge_dst,
                                      wt_rw1, rb1, wt_rw2, rb2, wt_rw3, rb3,
                                      wt_lw0, lb0, wt_lw1, alpha_dot,
                                      s_src, s_dst, v_src, v_dst,
                                      alpha, amax, val_s, m1v);

  k_denom<<<(kE * 4 + 255) / 256, 256, 0, stream>>>(alpha, edge_dst, amax, denom);

  k_aggregate<<<kE, 128, 0, stream>>>(alpha, edge_dst, denom, val_s, m1v,
                                      node_s, node_v);

  k_out_proj<<<kN / 16, 128, 0, stream>>>(node_s, node_v, wt_pw0, pb0, wt_pw1,
                                          (float*)d_out);
}
